// ALiBiMultiheadAttention_18494129177251
// MI455X (gfx1250) — matrix operand order repaired and checked by reading, not compiled, not run on hardware
//
#include <hip/hip_runtime.h>
#include <hip/hip_bf16.h>

// ---------------------------------------------------------------------------
// ALiBi multi-head attention for MI455X (gfx1250, wave32, WMMA bf16 + TDM)
// B=2, S=4096, E=512, H=8, Dk=64
// ---------------------------------------------------------------------------

#define B_   2
#define S_   4096
#define E_   512
#define H_   8
#define DK_  64
#define BH_  (B_ * H_)
#define M_   (B_ * S_)        // 8192 rows in the [B*S, E] GEMMs

typedef __attribute__((ext_vector_type(16))) __bf16 v16bf;
typedef __attribute__((ext_vector_type(8)))  __bf16 bf16x8;
typedef __attribute__((ext_vector_type(8)))  float  v8f;
typedef __attribute__((ext_vector_type(4)))  unsigned int u32x4;
typedef __attribute__((ext_vector_type(8)))  int i32x8;
typedef __attribute__((ext_vector_type(4)))  int i32x4;

// ---------------------------------------------------------------------------
// Fragment loaders (ISA 7.12.2 layouts, wave32)
// A (16-bit, 16x32, M x K): lane L holds row L&15, K chunks {hi*8..} and
// {16+hi*8..}, hi = L>=16. Memory row-major with given stride.
// ---------------------------------------------------------------------------
__device__ __forceinline__ v16bf load_a_bf16(const __bf16* base, int stride, int lane) {
    const int m  = lane & 15;
    const int hi = (lane >> 4) & 1;
    const __bf16* p = base + (size_t)m * stride + hi * 8;
    bf16x8 lo = *(const bf16x8*)(p);        // K = hi*8 .. hi*8+7
    bf16x8 hv = *(const bf16x8*)(p + 16);   // K = 16+hi*8 ..
    v16bf r;
#pragma unroll
    for (int i = 0; i < 8; ++i) { r[i] = lo[i]; r[8 + i] = hv[i]; }
    return r;
}

// B (16-bit, 32x16, K x N): lane L holds column N=L&15, K chunks {hi*8..} and
// {16+hi*8..}, hi = L>=16. Memory holds B^T row-major (column n contiguous along K).
__device__ __forceinline__ v16bf load_b_bf16(const __bf16* base, int stride, int lane) {
    const int n  = lane & 15;
    const int hi = (lane >> 4) & 1;
    const __bf16* p = base + (size_t)n * stride + hi * 8;
    bf16x8 lo = *(const bf16x8*)(p);
    bf16x8 hv = *(const bf16x8*)(p + 16);
    v16bf r;
#pragma unroll
    for (int i = 0; i < 8; ++i) { r[i] = lo[i]; r[8 + i] = hv[i]; }
    return r;
}

__device__ __forceinline__ v8f wmma_bf16(v16bf a, v16bf b, v8f c) {
    return __builtin_amdgcn_wmma_f32_16x16x32_bf16(false, a, false, b, (short)0, c,
                                                   false, false);
}

// ---------------------------------------------------------------------------
// Tensor Data Mover: 2D tile (bf16) global -> LDS, per ISA ch.8 D# layout.
// group0: [count=1 | lds_addr | global_addr | type=2]
// group1: [data_size=2B | tensor_dim0/1 | tile_dim0/1 | dim0 stride]
// groups 2/3 (and trailing group): zero (2D tensor).
// All args wave-uniform (SGPR descriptors).  6-arg builtin form (clang-23).
// ---------------------------------------------------------------------------
__device__ __forceinline__ void tdm_load_2d_bf16(unsigned lds_off,
                                                 const void* gaddr,
                                                 unsigned tile_d0, unsigned tile_d1,
                                                 unsigned tensor_d0, unsigned tensor_d1,
                                                 unsigned stride_d0) {
    const unsigned long long ga = (unsigned long long)(uintptr_t)gaddr;
    u32x4 g0;
    g0[0] = 1u;                                            // count=1, user mode
    g0[1] = lds_off;                                       // LDS byte address
    g0[2] = (unsigned)(ga & 0xFFFFFFFFu);                  // global_addr[31:0]
    g0[3] = (unsigned)((ga >> 32) & 0x1FFFFFFu) | (2u << 30); // addr[56:32]|type=2
    i32x8 g1;
    g1[0] = (int)(1u << 16);                               // data_size=1 (2 bytes)
    g1[1] = (int)((tensor_d0 & 0xFFFFu) << 16);            // tensor_dim0[15:0]
    g1[2] = (int)(((tensor_d0 >> 16) & 0xFFFFu) |
                  ((tensor_d1 & 0xFFFFu) << 16));          // td0[31:16] | td1[15:0]
    g1[3] = (int)(((tensor_d1 >> 16) & 0xFFFFu) |
                  ((tile_d0 & 0xFFFFu) << 16));            // td1[31:16] | tile_dim0
    g1[4] = (int)(tile_d1 & 0xFFFFu);                      // tile_dim1 (tile_dim2=0)
    g1[5] = (int)stride_d0;                                // dim0 stride low32
    g1[6] = 0;
    g1[7] = 0;
    const i32x4 z4 = {0, 0, 0, 0};
    const i32x8 z8 = {0, 0, 0, 0, 0, 0, 0, 0};
    __builtin_amdgcn_tensor_load_to_lds(g0, g1, z4, z4, z8, 0);
}

__device__ __forceinline__ unsigned lds_off_of(const void* p) {
    // Generic LDS addresses map addr[31:0] to the LDS byte offset (ISA 10.2).
    return (unsigned)(uintptr_t)p;
}

// ---------------------------------------------------------------------------
// Kernel 0: fp32 -> bf16 conversion of x and the four weight matrices
// ---------------------------------------------------------------------------
__global__ void cvt_bf16_kernel(const float* __restrict__ x,
                                const float* __restrict__ wq,
                                const float* __restrict__ wk,
                                const float* __restrict__ wv,
                                const float* __restrict__ wo,
                                __bf16* __restrict__ xb,
                                __bf16* __restrict__ wqb,
                                __bf16* __restrict__ wkb,
                                __bf16* __restrict__ wvb,
                                __bf16* __restrict__ wob) {
    const int NX = M_ * E_;
    const int NW = E_ * E_;
    for (int i = blockIdx.x * blockDim.x + threadIdx.x; i < NX;
         i += gridDim.x * blockDim.x) {
        xb[i] = (__bf16)x[i];
        if (i < NW) {
            wqb[i] = (__bf16)wq[i];
            wkb[i] = (__bf16)wk[i];
            wvb[i] = (__bf16)wv[i];
            wob[i] = (__bf16)wo[i];
        }
    }
}

// ---------------------------------------------------------------------------
// Kernel 1/3: projection GEMM  out[m,n] = sum_k A[m,k] * W[n,k] + bias[n]
// mode 0: Q -> scale 0.125, [BH,S,64]; mode 1: K -> [BH,S,64];
// mode 2: V -> transposed [BH,64,S]; mode 3: final projection -> f32 [M_,512]
// ---------------------------------------------------------------------------
__global__ void proj_gemm_kernel(const __bf16* __restrict__ A,
                                 const __bf16* __restrict__ W,
                                 const float*  __restrict__ bias,
                                 __bf16* __restrict__ out_bf,
                                 float*  __restrict__ out_f,
                                 int mode) {
    const int lane = threadIdx.x & 31;
    const int wave = threadIdx.x >> 5;
    const int hi   = (lane >> 4) & 1;
    const int nlo  = lane & 15;
    const int rowbase = blockIdx.x * 128 + wave * 16;

    const __bf16* arow = A + (size_t)rowbase * E_;

#pragma unroll 1
    for (int nt = 0; nt < E_ / 16; ++nt) {
        const int n0 = nt * 16;
        v8f acc;
        const float bval = bias[n0 + nlo];
#pragma unroll
        for (int r = 0; r < 8; ++r) acc[r] = bval;

#pragma unroll 4
        for (int ks = 0; ks < E_ / 32; ++ks) {
            v16bf af = load_a_bf16(arow + ks * 32, E_, lane);
            v16bf bf = load_b_bf16(W + (size_t)n0 * E_ + ks * 32, E_, lane);
            acc = wmma_bf16(af, bf, acc);
        }

        const int col = n0 + nlo;
#pragma unroll
        for (int r = 0; r < 8; ++r) {
            const int m = rowbase + r + 8 * hi;
            const float v = acc[r];
            if (mode == 3) {
                out_f[(size_t)m * E_ + col] = v;
            } else {
                const int b = m >> 12;
                const int s = m & (S_ - 1);
                const int h = col >> 6;
                const int d = col & (DK_ - 1);
                if (mode == 0) {
                    out_bf[(((size_t)(b * H_ + h)) * S_ + s) * DK_ + d] =
                        (__bf16)(v * 0.125f);   // fold 1/sqrt(Dk)
                } else if (mode == 1) {
                    out_bf[(((size_t)(b * H_ + h)) * S_ + s) * DK_ + d] = (__bf16)v;
                } else {
                    out_bf[(((size_t)(b * H_ + h)) * DK_ + d) * S_ + s] = (__bf16)v;
                }
            }
        }
    }
}

// ---------------------------------------------------------------------------
// Kernel 2: flash attention with ALiBi bias.
// Grid: (S/128, BH). Block 256 threads = 8 waves; wave owns 16 query rows.
// K/V tiles staged in LDS by the Tensor Data Mover, double-buffered so the
// DMA for tile i+1 overlaps compute on tile i (TENSORcnt pipelining).
// ---------------------------------------------------------------------------
__global__ void flash_alibi_kernel(const __bf16* __restrict__ qbf,   // [BH,S,64]
                                   const __bf16* __restrict__ kbf,   // [BH,S,64]
                                   const __bf16* __restrict__ vT,    // [BH,64,S]
                                   const unsigned char* __restrict__ mask, // [B,S]
                                   __bf16* __restrict__ attn) {      // [B*S, 512]
    __shared__ __bf16 p_lds[8 * 16 * 64];          // 16 KB: per-wave P tile
    __shared__ __bf16 kv_lds[2][2][64 * 64];       // 32 KB: [buf][K/V][64x64]

    const int lane = threadIdx.x & 31;
    const int wave = threadIdx.x >> 5;
    const int hi   = (lane >> 4) & 1;
    const int nlo  = lane & 15;

    const int bh = blockIdx.y;
    const int b  = bh >> 3;
    const int h  = bh & 7;
    const int q0 = blockIdx.x * 128 + wave * 16;

    const float slope = exp2f(-(float)(h + 1));    // ALiBi slope 2^-(h+1)

    const __bf16* qptr = qbf + (((size_t)bh) * S_ + q0) * DK_;
    const v16bf aq0 = load_a_bf16(qptr,      DK_, lane);
    const v16bf aq1 = load_a_bf16(qptr + 32, DK_, lane);

    float m_r[8], l_r[8];
    v8f   O[4];
#pragma unroll
    for (int r = 0; r < 8; ++r) { m_r[r] = -1e30f; l_r[r] = 0.0f; }
#pragma unroll
    for (int t = 0; t < 4; ++t)
#pragma unroll
        for (int r = 0; r < 8; ++r) O[t][r] = 0.0f;

    __bf16* pl = p_lds + wave * 16 * 64;
    const __bf16* kbase = kbf + ((size_t)bh) * S_ * DK_;
    const __bf16* vbase = vT + ((size_t)bh) * DK_ * S_;

    // Prologue: TDM-load tile 0 into buffer 0 (wave 0 drives the DMA).
    if (wave == 0) {
        tdm_load_2d_bf16(lds_off_of(&kv_lds[0][0][0]), kbase,
                         /*tile*/ 64, 64, /*tensor*/ DK_, S_, /*stride*/ DK_);
        tdm_load_2d_bf16(lds_off_of(&kv_lds[0][1][0]), vbase,
                         /*tile*/ 64, 64, /*tensor*/ S_, DK_, /*stride*/ S_);
    }

#pragma unroll 1
    for (int it = 0; it < S_ / 64; ++it) {
        const int j0  = it * 64;
        const int buf = it & 1;

        if (wave == 0) {
            if (it + 1 < S_ / 64) {
                const int jn = (it + 1) * 64;
                tdm_load_2d_bf16(lds_off_of(&kv_lds[buf ^ 1][0][0]),
                                 kbase + (size_t)jn * DK_,
                                 64, 64, DK_, S_, DK_);
                tdm_load_2d_bf16(lds_off_of(&kv_lds[buf ^ 1][1][0]),
                                 vbase + jn,
                                 64, 64, S_, DK_, S_);
                __builtin_amdgcn_s_wait_tensorcnt(2);  // tile `it` retired
            } else {
                __builtin_amdgcn_s_wait_tensorcnt(0);
            }
        }
        __syncthreads();                                // tile visible to all waves

        const __bf16* ktile = &kv_lds[buf][0][0];       // [64 keys][64 d]
        const __bf16* vtile = &kv_lds[buf][1][0];       // [64 d][64 keys]

        // ---- scores S = Q K^T (scale folded into Q) -----------------------
        v8f sfr[4];
#pragma unroll
        for (int t = 0; t < 4; ++t) {
            const __bf16* kp = ktile + (size_t)(t * 16) * DK_;
            v16bf bk0 = load_b_bf16(kp,      DK_, lane);
            v16bf bk1 = load_b_bf16(kp + 32, DK_, lane);
            v8f c;
#pragma unroll
            for (int r = 0; r < 8; ++r) c[r] = 0.0f;
            c = wmma_bf16(aq0, bk0, c);
            c = wmma_bf16(aq1, bk1, c);
            sfr[t] = c;
        }

        // ---- ALiBi bias + key padding mask --------------------------------
#pragma unroll
        for (int t = 0; t < 4; ++t) {
            const int j = j0 + t * 16 + nlo;
            const bool km = mask[(size_t)b * S_ + j] != 0;
#pragma unroll
            for (int r = 0; r < 8; ++r) {
                const int i = q0 + r + 8 * hi;
                float s = sfr[t][r] - slope * (float)(i - j);
                sfr[t][r] = km ? -1e30f : s;
            }
        }

        // ---- row max over tile (half-wave shfl reductions) ----------------
        float tmax[8];
#pragma unroll
        for (int r = 0; r < 8; ++r) {
            float m = fmaxf(fmaxf(sfr[0][r], sfr[1][r]),
                            fmaxf(sfr[2][r], sfr[3][r]));
#pragma unroll
            for (int off = 1; off < 16; off <<= 1)
                m = fmaxf(m, __shfl_xor(m, off, 32));
            tmax[r] = m;
        }

        // ---- online softmax update ----------------------------------------
        float scl[8], rsum[8];
#pragma unroll
        for (int r = 0; r < 8; ++r) {
            const float mn = fmaxf(m_r[r], tmax[r]);
            scl[r] = __expf(m_r[r] - mn);
            m_r[r] = mn;
            rsum[r] = 0.0f;
        }
#pragma unroll
        for (int t = 0; t < 4; ++t)
#pragma unroll
            for (int r = 0; r < 8; ++r) {
                const float p = __expf(sfr[t][r] - m_r[r]);
                sfr[t][r] = p;
                rsum[r] += p;
            }
#pragma unroll
        for (int r = 0; r < 8; ++r) {
            float s = rsum[r];
#pragma unroll
            for (int off = 1; off < 16; off <<= 1)
                s += __shfl_xor(s, off, 32);
            l_r[r] = l_r[r] * scl[r] + s;
        }
#pragma unroll
        for (int t = 0; t < 4; ++t)
#pragma unroll
            for (int r = 0; r < 8; ++r) O[t][r] *= scl[r];

        // ---- P: C-layout regs -> LDS (bf16) -> A-layout frags -------------
#pragma unroll
        for (int t = 0; t < 4; ++t)
#pragma unroll
            for (int r = 0; r < 8; ++r)
                pl[(r + 8 * hi) * 64 + t * 16 + nlo] = (__bf16)sfr[t][r];

        const v16bf ap0 = load_a_bf16(pl,      64, lane);   // keys 0..31
        const v16bf ap1 = load_a_bf16(pl + 32, 64, lane);   // keys 32..63

        // ---- O += P V (V tile in LDS, columns contiguous along keys) ------
#pragma unroll
        for (int nt = 0; nt < 4; ++nt) {
            const __bf16* vb = vtile + (size_t)(nt * 16) * 64;
            v16bf bv0 = load_b_bf16(vb,      64, lane);
            v16bf bv1 = load_b_bf16(vb + 32, 64, lane);
            O[nt] = wmma_bf16(ap0, bv0, O[nt]);
            O[nt] = wmma_bf16(ap1, bv1, O[nt]);
        }
        __syncthreads();    // all waves done with buf before TDM overwrites it
    }

    // ---- normalize and store attn output [B*S, E] (bf16) ------------------
    float inv[8];
#pragma unroll
    for (int r = 0; r < 8; ++r) inv[r] = 1.0f / fmaxf(l_r[r], 1e-30f);

#pragma unroll
    for (int t = 0; t < 4; ++t)
#pragma unroll
        for (int r = 0; r < 8; ++r) {
            const int grow = b * S_ + q0 + r + 8 * hi;
            const int col  = h * DK_ + t * 16 + nlo;
            attn[(size_t)grow * E_ + col] = (__bf16)(O[t][r] * inv[r]);
        }
}

// ---------------------------------------------------------------------------
// Host launcher
// ---------------------------------------------------------------------------
extern "C" void kernel_launch(void* const* d_in, const int* in_sizes, int n_in,
                              void* d_out, int out_size, void* d_ws, size_t ws_size,
                              hipStream_t stream) {
    const float* x  = (const float*)d_in[0];
    const unsigned char* kpm = (const unsigned char*)d_in[1]; // bool mask [B,S]
    const float* Wq = (const float*)d_in[2];
    const float* bq = (const float*)d_in[3];
    const float* Wk = (const float*)d_in[4];
    const float* bk = (const float*)d_in[5];
    const float* Wv = (const float*)d_in[6];
    const float* bv = (const float*)d_in[7];
    const float* Wo = (const float*)d_in[8];
    const float* bo = (const float*)d_in[9];
    float* out = (float*)d_out;

    char* ws = (char*)d_ws;
    size_t off = 0;
    __bf16* xb   = (__bf16*)(ws + off); off += (size_t)M_ * E_ * 2;
    __bf16* wqb  = (__bf16*)(ws + off); off += (size_t)E_ * E_ * 2;
    __bf16* wkb  = (__bf16*)(ws + off); off += (size_t)E_ * E_ * 2;
    __bf16* wvb  = (__bf16*)(ws + off); off += (size_t)E_ * E_ * 2;
    __bf16* wob  = (__bf16*)(ws + off); off += (size_t)E_ * E_ * 2;
    __bf16* qb   = (__bf16*)(ws + off); off += (size_t)BH_ * S_ * DK_ * 2;
    __bf16* kb   = (__bf16*)(ws + off); off += (size_t)BH_ * S_ * DK_ * 2;
    __bf16* vTb  = (__bf16*)(ws + off); off += (size_t)BH_ * DK_ * S_ * 2;
    __bf16* attn = (__bf16*)(ws + off); off += (size_t)M_ * E_ * 2;

    cvt_bf16_kernel<<<2048, 256, 0, stream>>>(x, Wq, Wk, Wv, Wo,
                                              xb, wqb, wkb, wvb, wob);

    proj_gemm_kernel<<<M_ / 128, 256, 0, stream>>>(xb, wqb, bq, qb,  nullptr, 0);
    proj_gemm_kernel<<<M_ / 128, 256, 0, stream>>>(xb, wkb, bk, kb,  nullptr, 1);
    proj_gemm_kernel<<<M_ / 128, 256, 0, stream>>>(xb, wvb, bv, vTb, nullptr, 2);

    flash_alibi_kernel<<<dim3(S_ / 128, BH_), 256, 0, stream>>>(qb, kb, vTb, kpm,
                                                                attn);

    proj_gemm_kernel<<<M_ / 128, 256, 0, stream>>>(attn, wob, bo, nullptr, out, 3);
}
